// concatDec_66279935312289
// MI455X (gfx1250) — matrix operand order repaired and checked by reading, not compiled, not run on hardware
//
#include <hip/hip_runtime.h>

// Edge-MLP decoder: y[e] = W2 . relu(W1 . concat(z[src[e]], z[dst[e]]) + b1) + b2
// N=50000, D=128 (2D=256), H=512, E=500000
//
// Strategy (MI455X / gfx1250):
//  - Stage all of W1 as f16 in LDS (512 rows x 264-half pitch = 264KB < 320KB/WGP).
//  - Each wave32 handles 32 edges/iter as two 16xK A-tiles held in VGPRs.
//  - 32 N-tiles x 8 K-chunks of v_wmma_f32_16x16x32_f16 per iteration,
//    issued as 4 independent accumulator chains to hide WMMA RAW latency.
//  - Epilogue: bias+relu, scale by W2, shfl_xor reduction over 16 lanes, store.

typedef __attribute__((ext_vector_type(16))) _Float16 v16h;
typedef __attribute__((ext_vector_type(8)))  _Float16 v8h;
typedef __attribute__((ext_vector_type(8)))  float    v8f;
typedef __attribute__((ext_vector_type(4)))  float    v4f;

static constexpr int kD     = 128;   // node feature dim
static constexpr int kTwoD  = 256;   // concat dim (K)
static constexpr int kH     = 512;   // hidden dim (N)
static constexpr int kPitch = 264;   // halves per LDS row of W1 (528B: 16B aligned, bank-spread)

// Load one 16-half B fragment (column = lane&15, K = 8*hf + [0,8) and 16 + 8*hf + [0,8)) from LDS.
__device__ __forceinline__ v16h load_bfrag(const _Float16* bp, int c) {
    const v8h blo = *(const v8h*)(bp + c * 32);
    const v8h bhi = *(const v8h*)(bp + c * 32 + 16);
    return __builtin_shufflevector(blo, bhi,
        0,1,2,3,4,5,6,7,8,9,10,11,12,13,14,15);
}

__device__ __forceinline__ v8f wmma_f16(v16h a, v16h b, v8f c) {
    return __builtin_amdgcn_wmma_f32_16x16x32_f16(
        false, a, false, b, (short)0, c, false, false);
}

__global__ __launch_bounds__(256)
void edge_mlp_wmma_kernel(const float* __restrict__ z,
                          const long long* __restrict__ eidx,   // [2, E] int64
                          const float* __restrict__ W1,         // [H, 2D]
                          const float* __restrict__ b1,         // [H]
                          const float* __restrict__ W2,         // [1, H]
                          const float* __restrict__ b2,         // [1]
                          float* __restrict__ out,              // [E]
                          int E, int nIter)
{
    extern __shared__ char smem[];
    _Float16* w1s = (_Float16*)smem;                                   // [512][264] halves
    float2*   bws = (float2*)(smem + (size_t)kH * kPitch * sizeof(_Float16)); // {b1,W2}[512]

    const int tid = threadIdx.x;

    // ---- Stage W1 (f32 -> f16), {b1, W2} pairs into LDS ----
    for (int i = tid; i < kH * kTwoD; i += 256) {
        const int row = i >> 8;          // W1 row = output column
        const int col = i & 255;         // K index
        w1s[row * kPitch + col] = (_Float16)W1[i];
    }
    for (int i = tid; i < kH; i += 256) {
        bws[i] = make_float2(b1[i], W2[i]);
    }
    __syncthreads();

    const int lane = tid & 31;
    const int wave = tid >> 5;
    const int hf   = lane >> 4;     // 0: lanes 0-15, 1: lanes 16-31
    const int l16  = lane & 15;
    const float b2v = b2[0];

    const int waveGlobal = blockIdx.x * 8 + wave;
    const int nWaves     = gridDim.x * 8;

    for (int it = waveGlobal; it < nIter; it += nWaves) {
        const long long base = (long long)it * 32;

        // ---- Load A fragments: two 16-edge tiles, 8 K-chunks of 32 ----
        // 16-bit A 16x32 layout: lane holds K = 8*hf+{0..7} and 16+8*hf+{0..7}.
        v16h afrag[2][8];
        #pragma unroll
        for (int t = 0; t < 2; ++t) {
            long long e = base + t * 16 + l16;
            if (e >= E) e = E - 1;                 // clamp (keeps EXEC full for WMMA)
            const long long s = eidx[e];
            const long long d = eidx[(long long)E + e];
            const float* zs = z + s * kD;
            const float* zd = z + d * kD;
            #pragma unroll
            for (int c = 0; c < 8; ++c) {
                const float* zp = (c < 4 ? zs : zd) + (c & 3) * 32 + hf * 8;
                const v4f f0 = *(const v4f*)(zp +  0);
                const v4f f1 = *(const v4f*)(zp +  4);
                const v4f f2 = *(const v4f*)(zp + 16);
                const v4f f3 = *(const v4f*)(zp + 20);
                v16h a;
                #pragma unroll
                for (int j = 0; j < 4; ++j) {
                    a[j]      = (_Float16)f0[j];
                    a[4 + j]  = (_Float16)f1[j];
                    a[8 + j]  = (_Float16)f2[j];
                    a[12 + j] = (_Float16)f3[j];
                }
                afrag[t][c] = a;
            }
        }

        float y0[8], y1[8];
        #pragma unroll
        for (int r = 0; r < 8; ++r) { y0[r] = 0.0f; y1[r] = 0.0f; }

        // ---- 32 N-tiles over H=512 ----
        for (int n = 0; n < 32; ++n) {
            const int ncol = n * 16 + l16;         // output column handled by this lane
            const _Float16* bp = w1s + ncol * kPitch + hf * 8;

            // 4 independent accumulator chains (2 tiles x 2 K-halves):
            // 3 unrelated WMMAs between any RAW-dependent pair.
            v8f a00 = {}, a01 = {}, a10 = {}, a11 = {};
            #pragma unroll
            for (int c = 0; c < 4; ++c) {
                const v16h bLo = load_bfrag(bp, c);
                const v16h bHi = load_bfrag(bp, c + 4);
                a00 = wmma_f16(afrag[0][c],     bLo, a00);
                a10 = wmma_f16(afrag[1][c],     bLo, a10);
                a01 = wmma_f16(afrag[0][c + 4], bHi, a01);
                a11 = wmma_f16(afrag[1][c + 4], bHi, a11);
            }

            const float2 bw = bws[ncol];           // one ds_load_b64: {bias, w2}
            #pragma unroll
            for (int r = 0; r < 8; ++r) {
                float h0 = a00[r] + a01[r] + bw.x;
                float h1 = a10[r] + a11[r] + bw.x;
                h0 = h0 > 0.0f ? h0 : 0.0f;        // plain select: single v_max
                h1 = h1 > 0.0f ? h1 : 0.0f;
                y0[r] += h0 * bw.y;
                y1[r] += h1 * bw.y;
            }
        }

        // ---- Reduce partial y across the 16 column-lanes (stays within each half) ----
        #pragma unroll
        for (int r = 0; r < 8; ++r) {
            #pragma unroll
            for (int m = 8; m >= 1; m >>= 1) {
                y0[r] += __shfl_xor(y0[r], m, 32);
                y1[r] += __shfl_xor(y1[r], m, 32);
            }
        }

        // C/D layout: VGPR r <-> row (r + 8*hf); lane l16==0 holds the full sums.
        if (l16 == 0) {
            const long long e0 = base + hf * 8;        // tile 0 rows
            const long long e1 = base + 16 + hf * 8;   // tile 1 rows
            #pragma unroll
            for (int r = 0; r < 8; ++r) {
                if (e0 + r < E) out[e0 + r] = y0[r] + b2v;
                if (e1 + r < E) out[e1 + r] = y1[r] + b2v;
            }
        }
    }
}

extern "C" void kernel_launch(void* const* d_in, const int* in_sizes, int n_in,
                              void* d_out, int out_size, void* d_ws, size_t ws_size,
                              hipStream_t stream) {
    const float*     z    = (const float*)d_in[0];
    const long long* eidx = (const long long*)d_in[1];
    const float*     W1   = (const float*)d_in[2];
    const float*     b1   = (const float*)d_in[3];
    const float*     W2   = (const float*)d_in[4];
    const float*     b2   = (const float*)d_in[5];
    float*           out  = (float*)d_out;

    const int E     = in_sizes[1] / 2;          // edge_index is [2, E]
    const int nIter = (E + 31) / 32;            // 32 edges per wave-iteration

    const size_t shmem = (size_t)kH * kPitch * sizeof(_Float16)   // W1 f16
                       + (size_t)kH * sizeof(float2);             // {b1, W2} pairs
    // >64KB dynamic LDS: raise the limit (gfx1250 WGP has 320KB).
    hipFuncSetAttribute((const void*)edge_mlp_wmma_kernel,
                        hipFuncAttributeMaxDynamicSharedMemorySize, (int)shmem);

    const int grid = 256;   // grid-stride persistent waves; 8 waves per WG
    edge_mlp_wmma_kernel<<<grid, 256, shmem, stream>>>(z, eidx, W1, b1, W2, b2,
                                                       out, E, nIter);
}
